// TfModel_57140244906038
// MI455X (gfx1250) — compile-verified, matrix operand order checked by reading, not run on hardware
//
#include <hip/hip_runtime.h>

// ---------------------------------------------------------------------------
// MI455X (gfx1250) implementation.
//  - wave32, WMMA 16x16x32 f16->f32 for all GEMMs (A@W^T form: both fragments
//    are contiguous 32B/lane loads, same k = 16*(lane/16)+s order for A and B).
//  - Decoder scan (sequential, latency-bound) runs as ONE persistent 1024-thread
//    workgroup with h resident in LDS. Per-step 32KB weights are double-buffered
//    into LDS with the Tensor Data Mover (tensor_load_to_lds, TENSORcnt); all
//    per-step bias/LN vectors are staged once via GLOBAL_LOAD_ASYNC_TO_LDS_B128
//    (ASYNCcnt). ~316KB of the 320KB WGP LDS is used.
//  - Cross-attn K/V projections hoisted out of the 400-step loop (the reference
//    recomputes them every step: ~430 GFLOP of the ~434 GFLOP total).
// ---------------------------------------------------------------------------

typedef __attribute__((ext_vector_type(16))) _Float16 v16h;
typedef __attribute__((ext_vector_type(8)))  float    v8f;
typedef unsigned int u32;

#define DEV __device__ __forceinline__

DEV v8f wmma16(v16h a, v16h b, v8f c) {
  // (neg_a, A, neg_b, B, c_mod, C, reuse_a, reuse_b)
  return __builtin_amdgcn_wmma_f32_16x16x32_f16(false, a, false, b, (short)0, c,
                                                false, false);
}

// ----- CDNA5 async-LDS helpers (CDNA5 ISA) -----------
// One 16B chunk global -> LDS, tracked by ASYNCcnt. LDS destination address =
// low 32 bits of the generic pointer (flat->LDS truncation, ISA §10.2).
DEV void async_ld_b128(const void* gaddr, void* lds_ptr) {
  unsigned int laddr = (unsigned int)(unsigned long long)lds_ptr;
  unsigned long long ga = (unsigned long long)gaddr;
  asm volatile("global_load_async_to_lds_b128 %0, %1, off"
               :: "v"(laddr), "v"(ga)
               : "memory");
}
DEV void wait_asynccnt0() {
  asm volatile("s_wait_asynccnt 0x0" ::: "memory");
}

// ----- CDNA5 Tensor Data Mover (CDNA5 ISA-8) ---------
// One TDM descriptor moves a whole 128x128 f16 (32KB) weight tile to LDS.
// 2D tensor of 8-byte elements: dim0 = 32 elems/row, dim1 = 128 rows.
// D# group0: count=1 | lds_addr | global_addr | type=2.
// D# group1: data_size=8B, tensor_dim0=tile_dim0=32, tensor_dim1=tile_dim1=128,
// dim0_stride=32.  The descriptor must live in SGPR groups: runtime dwords are
// moved into fixed SGPRs with v_readfirstlane (values are wave-uniform), the
// constant dwords with s_mov, then tensor_load_to_lds s[80:83], s[84:91].
DEV void tdm_load_weight(const void* gsrc, void* ldsDst) {
  unsigned long long ga = (unsigned long long)gsrc;
  u32 laddr = (u32)(unsigned long long)ldsDst;
  u32 galo = (u32)ga;
  u32 gahi = (u32)((ga >> 32) & 0x01FFFFFFu) | (2u << 30);  // addr[56:32] | type=2
  asm volatile(
      "s_mov_b32 s80, 1\n\t"                 // g0[0]: count=1 (valid user D#)
      "v_readfirstlane_b32 s81, %0\n\t"      // g0[1]: lds_addr
      "v_readfirstlane_b32 s82, %1\n\t"      // g0[2]: global_addr[31:0]
      "v_readfirstlane_b32 s83, %2\n\t"      // g0[3]: global_addr[56:32] | type=2
      "s_mov_b32 s84, 0x30000\n\t"           // g1[0]: data_size=3 (8B)
      "s_mov_b32 s85, 0x200000\n\t"          // g1[1]: tensor_dim0=32  (<<16)
      "s_mov_b32 s86, 0x800000\n\t"          // g1[2]: tensor_dim1=128 (<<16)
      "s_mov_b32 s87, 0x200000\n\t"          // g1[3]: tile_dim0=32    (<<16)
      "s_mov_b32 s88, 128\n\t"               // g1[4]: tile_dim1=128, tile_dim2=0
      "s_mov_b32 s89, 32\n\t"                // g1[5]: tensor_dim0_stride=32
      "s_mov_b32 s90, 0\n\t"                 // g1[6]
      "s_mov_b32 s91, 0\n\t"                 // g1[7]
      "tensor_load_to_lds s[80:83], s[84:91]"
      :: "v"(laddr), "v"(galo), "v"(gahi)
      : "memory", "s80", "s81", "s82", "s83", "s84", "s85",
        "s86", "s87", "s88", "s89", "s90", "s91");
}

// ----- model constants ------------------------------------------------------
#define DMODEL   128
#define ENC_B    64     // encoder batch
#define ENC_L    128    // encoder sequence length
#define NHEADC   2
#define HD       64
#define DEC_B    128    // decoder batch (== L_SRC due to reference's transpose)
#define MEM_L    64     // decoder memory length (== B_SRC)
#define TSTEPS   400
#define FC_OUTC  2048
#define NMELC    80
#define NSTFTC   512
#define KPAD     96     // 80 padded to multiple of 32 for the WMMA K loop
#define ENC_ROWS (ENC_B * ENC_L)          // 8192
#define EPI_ROWS (DEC_B * TSTEPS)         // 51200

// ----- small utility kernels ------------------------------------------------

__global__ void pack_f16_kernel(const float* __restrict__ src, _Float16* __restrict__ dst,
                                int rows, int cols, int dstld) {
  int idx = blockIdx.x * blockDim.x + threadIdx.x;
  if (idx >= rows * dstld) return;
  int r = idx / dstld, c = idx - r * dstld;
  dst[idx] = (c < cols) ? (_Float16)src[(size_t)r * cols + c] : (_Float16)0.0f;
}

__global__ void pack_melT_kernel(const float* __restrict__ fb, _Float16* __restrict__ mt) {
  // fb: (512,80) f32  ->  mt: (80,512) f16  (so B-fragment loads are contiguous)
  int idx = blockIdx.x * blockDim.x + threadIdx.x;
  if (idx >= NMELC * NSTFTC) return;
  int f = idx & (NSTFTC - 1);
  int m = idx >> 9;
  mt[idx] = (_Float16)fb[(size_t)f * NMELC + m];
}

__global__ void zero_f16_kernel(_Float16* p, int n) {
  int idx = blockIdx.x * blockDim.x + threadIdx.x;
  if (idx < n) p[idx] = (_Float16)0.0f;
}

__global__ void embed_pe_kernel(const int* __restrict__ src, const float* __restrict__ emb,
                                float* __restrict__ x, _Float16* __restrict__ xh) {
  // x[b][l][d] = emb[src[l][b]][d] + pe[l][d]
  int idx = blockIdx.x * blockDim.x + threadIdx.x;
  if (idx >= ENC_B * ENC_L * DMODEL) return;
  int d = idx & 127;
  int l = (idx >> 7) & 127;
  int b = idx >> 14;
  int tok = src[l * ENC_B + b];
  float freq = __expf(-9.210340371976184f * (float)(d & ~1) / 128.0f);
  float ang = (float)l * freq;
  float pe = (d & 1) ? __cosf(ang) : __sinf(ang);
  float v = emb[(size_t)tok * DMODEL + d] + pe;
  x[idx] = v;
  xh[idx] = (_Float16)v;
}

__global__ void resid_ln_kernel(float* __restrict__ x, const float* __restrict__ t,
                                const float* __restrict__ g, const float* __restrict__ be,
                                _Float16* __restrict__ xh, int rows) {
  int row  = blockIdx.x * (blockDim.x >> 5) + (threadIdx.x >> 5);
  int lane = threadIdx.x & 31;
  if (row >= rows) return;
  float v[4], s = 0.0f;
#pragma unroll
  for (int j = 0; j < 4; ++j) {
    int c = lane + 32 * j;
    v[j] = x[(size_t)row * DMODEL + c] + t[(size_t)row * DMODEL + c];
    s += v[j];
  }
#pragma unroll
  for (int o = 16; o > 0; o >>= 1) s += __shfl_xor(s, o, 32);
  float mu = s * (1.0f / 128.0f);
  float q = 0.0f;
#pragma unroll
  for (int j = 0; j < 4; ++j) { float d = v[j] - mu; q += d * d; }
#pragma unroll
  for (int o = 16; o > 0; o >>= 1) q += __shfl_xor(q, o, 32);
  float inv = rsqrtf(q * (1.0f / 128.0f) + 1e-5f);
#pragma unroll
  for (int j = 0; j < 4; ++j) {
    int c = lane + 32 * j;
    float ov = (v[j] - mu) * inv * g[c] + be[c];
    x[(size_t)row * DMODEL + c] = ov;
    xh[(size_t)row * DMODEL + c] = (_Float16)ov;
  }
}

__global__ void softmax_rows128_kernel(const float* __restrict__ s, _Float16* __restrict__ p,
                                       int rows) {
  int row  = blockIdx.x * (blockDim.x >> 5) + (threadIdx.x >> 5);
  int lane = threadIdx.x & 31;
  if (row >= rows) return;
  float v[4], mx = -3.0e38f;
#pragma unroll
  for (int j = 0; j < 4; ++j) { v[j] = s[(size_t)row * 128 + lane + 32 * j]; mx = fmaxf(mx, v[j]); }
#pragma unroll
  for (int o = 16; o > 0; o >>= 1) mx = fmaxf(mx, __shfl_xor(mx, o, 32));
  float sum = 0.0f;
#pragma unroll
  for (int j = 0; j < 4; ++j) { v[j] = __expf(v[j] - mx); sum += v[j]; }
#pragma unroll
  for (int o = 16; o > 0; o >>= 1) sum += __shfl_xor(sum, o, 32);
  float inv = 1.0f / sum;
#pragma unroll
  for (int j = 0; j < 4; ++j) p[(size_t)row * 128 + lane + 32 * j] = (_Float16)(v[j] * inv);
}

__global__ void transpose_vT_kernel(const _Float16* __restrict__ vp, _Float16* __restrict__ vT) {
  // encoder: vp (64,128,128) -> vT [(b*2+h)][d][m]  (m over L=128)
  int idx = blockIdx.x * blockDim.x + threadIdx.x;
  if (idx >= ENC_B * NHEADC * HD * ENC_L) return;
  int m = idx & 127;
  int d = (idx >> 7) & 63;
  int z = idx >> 13;            // b*2 + h
  int b = z >> 1, h = z & 1;
  vT[idx] = vp[((size_t)b * ENC_L + m) * DMODEL + h * HD + d];
}

__global__ void transpose_caVT_kernel(const _Float16* __restrict__ vp, _Float16* __restrict__ vT) {
  // decoder cross V: vp is (benc=64, l=128, 128); memory[bdec][m][d] = vp[m][bdec][d]
  // vT[((bdec*2+hd)*64 + d)*64 + m]
  int idx = blockIdx.x * blockDim.x + threadIdx.x;
  if (idx >= DEC_B * NHEADC * HD * MEM_L) return;
  int m  = idx & 63;
  int d  = (idx >> 6) & 63;
  int z  = idx >> 12;           // bdec*2 + hd
  int hd = z & 1, bdec = z >> 1;
  vT[idx] = vp[((size_t)m * ENC_L + bdec) * DMODEL + hd * HD + d];
}

__global__ void lips_kernel(const float* __restrict__ pres, const float* __restrict__ vel,
                            const float* __restrict__ Am, const float* __restrict__ Bm,
                            _Float16* __restrict__ lipsH, int total) {
  int idx = blockIdx.x * blockDim.x + threadIdx.x;
  if (idx >= total) return;
  lipsH[idx] = (_Float16)(Am[idx] * pres[idx] + Bm[idx] * vel[idx]);
}

// ----- generic batched WMMA GEMM:  C = act(scale * A @ W^T + bias) ---------
__global__ void gemm_wmma(const _Float16* __restrict__ A, long long sA1, long long sA2, int lda,
                          const _Float16* __restrict__ W, long long sW1, long long sW2, int ldw,
                          const float* __restrict__ bias,
                          float* __restrict__ C, long long sC1, long long sC2, int ldc,
                          _Float16* __restrict__ Ch, long long sCh1, long long sCh2, int ldch,
                          int M, int N, int K, int H, float scale, int act) {
  int bz = blockIdx.z;
  int b1 = bz / H, b2 = bz - b1 * H;
  A += b1 * sA1 + b2 * sA2;
  W += b1 * sW1 + b2 * sW2;
  if (C)  C  += b1 * sC1  + b2 * sC2;
  if (Ch) Ch += b1 * sCh1 + b2 * sCh2;

  int wave = threadIdx.x >> 5, lane = threadIdx.x & 31;
  int tilesN = N >> 4;
  int tile = blockIdx.x * (blockDim.x >> 5) + wave;
  if (tile >= (M >> 4) * tilesN) return;          // wave-uniform exit (EXEC stays all-1)
  int tmi = tile / tilesN;
  int tm = tmi << 4, tn = (tile - tmi * tilesN) << 4;

  int mr = lane & 15, kb = (lane >> 4) << 4;
  const _Float16* ap = A + (size_t)(tm + mr) * lda + kb;
  const _Float16* wp = W + (size_t)(tn + mr) * ldw + kb;

  v8f acc = {0.f, 0.f, 0.f, 0.f, 0.f, 0.f, 0.f, 0.f};
  for (int k0 = 0; k0 < K; k0 += 32) {
    __builtin_prefetch(ap + k0 + 64, 0, 1);       // -> global_prefetch_b8
    __builtin_prefetch(wp + k0 + 64, 0, 1);
    v16h av = *(const v16h*)(ap + k0);
    v16h wv = *(const v16h*)(wp + k0);
    acc = wmma16(av, wv, acc);
  }

  int nc = lane & 15, mh = (lane >> 4) << 3;
#pragma unroll
  for (int r = 0; r < 8; ++r) {
    float v = acc[r] * scale;
    if (bias) v += bias[tn + nc];
    if (act)  v = fmaxf(v, 0.0f);
    size_t row = (size_t)(tm + mh + r);
    if (C)  C[row * ldc + tn + nc] = v;
    if (Ch) Ch[row * ldch + tn + nc] = (_Float16)v;
  }
}

// ----- fused persistent decoder --------------------------------------------

struct DecLayerGPU {
  const _Float16 *sa_wv, *sa_wo, *ca_wq, *ca_wo, *w1, *w2;   // f16 weights (N,K)
  const float *sa_bv, *sa_bo, *ca_bq, *ca_bo, *b1, *b2;
  const float *g1, *be1, *g2, *be2, *g3, *be3;
  const _Float16 *caK;   // (benc=64, l=128, 128) f16  (memory K projection)
  const _Float16 *caVT;  // (bdec,hd,d,m) f16, m over 64
};
struct DecArgs {
  DecLayerGPU L[2];
  const _Float16* fcW;   // (2048,128) f16
  const float *fcB, *gateW, *gateB;
  float *pres, *vel, *Am, *Bm, *gate;
  int T;
};

// 128x128 GEMM: A (LDS) @ W^T (LDS, TDM-staged); 64 tiles over 32 waves.
DEV void gemm_lds(const _Float16* __restrict__ A, const _Float16* __restrict__ W,
                  const float* __restrict__ bias, float* outF, _Float16* outH,
                  bool relu, int wave, int lane) {
  int mr = lane & 15, kb = (lane >> 4) << 4;
  int nc = lane & 15, mh = (lane >> 4) << 3;
  for (int ti = wave; ti < 64; ti += 32) {
    int tm = (ti >> 3) << 4, tn = (ti & 7) << 4;
    v8f acc = {0.f, 0.f, 0.f, 0.f, 0.f, 0.f, 0.f, 0.f};
    const _Float16* ap = A + (tm + mr) * DMODEL + kb;   // LDS -> ds_read_b128
    const _Float16* wp = W + (tn + mr) * DMODEL + kb;   // LDS -> ds_read_b128
#pragma unroll
    for (int k0 = 0; k0 < DMODEL; k0 += 32) {
      v16h av = *(const v16h*)(ap + k0);
      v16h bv = *(const v16h*)(wp + k0);
      acc = wmma16(av, bv, acc);
    }
#pragma unroll
    for (int r = 0; r < 8; ++r) {
      float v = acc[r] + bias[tn + nc];
      if (relu) v = fmaxf(v, 0.0f);
      int o = (tm + mh + r) * DMODEL + tn + nc;
      if (outF) outF[o] = v;
      if (outH) outH[o] = (_Float16)v;
    }
  }
}

// h = LN(h + t); 128 rows over 32 waves (4 rows/wave).
DEV void ln_update(float* h, const float* t, const float* g, const float* be,
                   _Float16* hh, int wave, int lane) {
  for (int rr = 0; rr < 4; ++rr) {
    int row = wave * 4 + rr;
    float v[4], s = 0.0f;
#pragma unroll
    for (int j = 0; j < 4; ++j) {
      int c = lane + 32 * j;
      v[j] = h[row * DMODEL + c] + t[row * DMODEL + c];
      s += v[j];
    }
#pragma unroll
    for (int o = 16; o > 0; o >>= 1) s += __shfl_xor(s, o, 32);
    float mu = s * (1.0f / 128.0f), q = 0.0f;
#pragma unroll
    for (int j = 0; j < 4; ++j) { float d = v[j] - mu; q += d * d; }
#pragma unroll
    for (int o = 16; o > 0; o >>= 1) q += __shfl_xor(q, o, 32);
    float inv = rsqrtf(q * (1.0f / 128.0f) + 1e-5f);
#pragma unroll
    for (int j = 0; j < 4; ++j) {
      int c = lane + 32 * j;
      float ov = (v[j] - mu) * inv * g[c] + be[c];
      h[row * DMODEL + c] = ov;
      hh[row * DMODEL + c] = (_Float16)ov;
    }
  }
}

// LDS constant-bank slot indices (128 floats each, per layer)
#define CB_SAV 0
#define CB_SAO 1
#define CB_CAQ 2
#define CB_CAO 3
#define CB_B1  4
#define CB_B2  5
#define CB_G1  6
#define CB_BE1 7
#define CB_G2  8
#define CB_BE2 9
#define CB_G3  10
#define CB_BE3 11
#define CB_FCB  3072   // 2048 floats
#define CB_GATE 5120   // 128 floats

__global__ void __launch_bounds__(1024) decoder_scan(DecArgs da) {
  extern __shared__ char smem_raw[];
  float*    sh_h  = (float*)smem_raw;              // 16384 f32 : h (128x128)
  float*    sh_t  = sh_h + 16384;                  // 16384 f32 : temps / scores
  _Float16* sh_hh = (_Float16*)(sh_t + 16384);     // 16384 f16 : h
  _Float16* sh_p  = sh_hh + 16384;                 // 16384 f16 : probs
  _Float16* sh_b  = sh_p + 16384;                  // 16384 f16 : temp (q / v / o / ffn)
  _Float16* sh_w0 = sh_b + 16384;                  // 16384 f16 : weight buffer 0
  _Float16* sh_w1 = sh_w0 + 16384;                 // 16384 f16 : weight buffer 1
  float*    sh_c  = (float*)(sh_w1 + 16384);       //  5376 f32 : biases/LN/fcB/gateW

  int tid = threadIdx.x, wave = tid >> 5, lane = tid & 31;

  for (int i = tid; i < 16384; i += 1024) { sh_h[i] = 0.0f; sh_hh[i] = (_Float16)0.0f; }

  // ---- one-time async-LDS staging of all per-step float vectors ----
  {
    const float* vecs[24] = {
      da.L[0].sa_bv, da.L[0].sa_bo, da.L[0].ca_bq, da.L[0].ca_bo, da.L[0].b1, da.L[0].b2,
      da.L[0].g1, da.L[0].be1, da.L[0].g2, da.L[0].be2, da.L[0].g3, da.L[0].be3,
      da.L[1].sa_bv, da.L[1].sa_bo, da.L[1].ca_bq, da.L[1].ca_bo, da.L[1].b1, da.L[1].b2,
      da.L[1].g1, da.L[1].be1, da.L[1].g2, da.L[1].be2, da.L[1].g3, da.L[1].be3 };
    for (int q = tid; q < 24 * 32; q += 1024)       // 24 vectors x 32 16B chunks
      async_ld_b128(vecs[q >> 5] + (q & 31) * 4, sh_c + (q >> 5) * 128 + (q & 31) * 4);
    for (int q = tid; q < 512; q += 1024)           // fcB: 2048 f32 = 512 chunks
      async_ld_b128(da.fcB + q * 4, sh_c + CB_FCB + q * 4);
    if (tid < 32)                                   // gateW: 128 f32 = 32 chunks
      async_ld_b128(da.gateW + tid * 4, sh_c + CB_GATE + tid * 4);
    wait_asynccnt0();
  }
  __syncthreads();

  float gb = da.gateB[0];

  // per-step weight streaming order (matches the gemm sequence below)
  const _Float16* wlist[12] = {
    da.L[0].sa_wv, da.L[0].sa_wo, da.L[0].ca_wq, da.L[0].ca_wo, da.L[0].w1, da.L[0].w2,
    da.L[1].sa_wv, da.L[1].sa_wo, da.L[1].ca_wq, da.L[1].ca_wo, da.L[1].w1, da.L[1].w2 };

  for (int t = 0; t < da.T; ++t) {
    int gidx = 0;
    if (wave == 0) tdm_load_weight(wlist[0], sh_w0);  // kick off first weight DMA

    // wave0: issue next weight's TDM, wait for current (in-order per wave),
    // then the whole workgroup consumes the staged buffer.
    auto staged_gemm = [&](const _Float16* Aptr, const float* bias,
                           float* outF, _Float16* outH, bool relu) {
      if (wave == 0) {
        if (gidx + 1 < 12) {
          tdm_load_weight(wlist[gidx + 1], (gidx & 1) ? sh_w0 : sh_w1);
          __builtin_amdgcn_s_wait_tensorcnt(1);       // current buffer complete
        } else {
          __builtin_amdgcn_s_wait_tensorcnt(0);
        }
      }
      __syncthreads();
      gemm_lds(Aptr, (gidx & 1) ? sh_w1 : sh_w0, bias, outF, outH, relu, wave, lane);
      __syncthreads();
      ++gidx;
    };

    for (int l = 0; l < 2; ++l) {
      const DecLayerGPU& P = da.L[l];
      const float* cb = sh_c + l * 12 * 128;

      // --- self-attn over length-1 sequence: softmax==1, collapses exactly ---
      staged_gemm(sh_hh, cb + CB_SAV * 128, nullptr, sh_b, false);
      staged_gemm(sh_b,  cb + CB_SAO * 128, sh_t, nullptr, false);
      ln_update(sh_h, sh_t, cb + CB_G1 * 128, cb + CB_BE1 * 128, sh_hh, wave, lane);
      __syncthreads();

      // --- cross-attn (memory length 64) ---
      staged_gemm(sh_hh, cb + CB_CAQ * 128, nullptr, sh_b, false);   // q
      // scores: sh_t[(bdec*2+hd)*64 + m]
      for (int i = tid; i < DEC_B * NHEADC * MEM_L; i += 1024) {
        int m = i & 63, z = i >> 6;
        int bd = z >> 1, hd = z & 1;
        const _Float16* q  = sh_b + bd * DMODEL + hd * HD;
        const _Float16* kk = P.caK + ((size_t)m * ENC_L + bd) * DMODEL + hd * HD;
        float s = 0.0f;
        for (int j = 0; j < HD; ++j) s += (float)q[j] * (float)kk[j];
        sh_t[i] = s * 0.125f;
      }
      __syncthreads();
      // softmax over 256 rows of 64 -> sh_p
      for (int rr = 0; rr < 8; ++rr) {
        int row = wave * 8 + rr;
        float a0 = sh_t[row * 64 + lane];
        float a1 = sh_t[row * 64 + lane + 32];
        float mx = fmaxf(a0, a1);
        for (int o = 16; o > 0; o >>= 1) mx = fmaxf(mx, __shfl_xor(mx, o, 32));
        float e0 = __expf(a0 - mx), e1 = __expf(a1 - mx);
        float s = e0 + e1;
        for (int o = 16; o > 0; o >>= 1) s += __shfl_xor(s, o, 32);
        float inv = 1.0f / s;
        sh_p[row * 64 + lane]      = (_Float16)(e0 * inv);
        sh_p[row * 64 + lane + 32] = (_Float16)(e1 * inv);
      }
      __syncthreads();
      // o = probs @ V : sh_b[bd*128 + hd*64 + d]
      for (int i = tid; i < DEC_B * DMODEL; i += 1024) {
        int d = i & 63, hd = (i >> 6) & 1, bd = i >> 7;
        const _Float16* pr = sh_p + (bd * 2 + hd) * 64;
        const _Float16* vv = P.caVT + ((size_t)(bd * 2 + hd) * HD + d) * MEM_L;
        float s = 0.0f;
        for (int m = 0; m < MEM_L; ++m) s += (float)pr[m] * (float)vv[m];
        sh_b[bd * DMODEL + hd * HD + d] = (_Float16)s;
      }
      __syncthreads();
      staged_gemm(sh_b, cb + CB_CAO * 128, sh_t, nullptr, false);
      ln_update(sh_h, sh_t, cb + CB_G2 * 128, cb + CB_BE2 * 128, sh_hh, wave, lane);
      __syncthreads();

      // --- FFN ---
      staged_gemm(sh_hh, cb + CB_B1 * 128, nullptr, sh_b, true);
      staged_gemm(sh_b,  cb + CB_B2 * 128, sh_t, nullptr, false);
      ln_update(sh_h, sh_t, cb + CB_G3 * 128, cb + CB_BE3 * 128, sh_hh, wave, lane);
      __syncthreads();
    }

    // --- FC 128x2048 (512KB weight stays on the global-fragment path),
    //     scattered into pres/vel/A/Bm in (B,T,512) layout ---
    {
      int mr = lane & 15, kb = (lane >> 4) << 4;
      int nc = lane & 15, mh = (lane >> 4) << 3;
      for (int ti = wave; ti < (DEC_B / 16) * (FC_OUTC / 16); ti += 32) {
        int tm = (ti >> 7) << 4;
        int tn = (ti & 127) << 4;
        v8f acc = {0.f, 0.f, 0.f, 0.f, 0.f, 0.f, 0.f, 0.f};
        const _Float16* ap = sh_hh + (tm + mr) * DMODEL + kb;
        const _Float16* wp = da.fcW + (size_t)(tn + mr) * DMODEL + kb;
#pragma unroll
        for (int k0 = 0; k0 < DMODEL; k0 += 32) {
          __builtin_prefetch(wp + k0 + 64, 0, 1);
          v16h av = *(const v16h*)(ap + k0);
          v16h wv = *(const v16h*)(wp + k0);
          acc = wmma16(av, wv, acc);
        }
        int region = tn >> 9;     // uniform within a tile (512 % 16 == 0)
        float* base = region == 0 ? da.pres : region == 1 ? da.vel
                    : region == 2 ? da.Am   : da.Bm;
#pragma unroll
        for (int r = 0; r < 8; ++r) {
          int b = tm + mh + r, col = tn + nc;
          float v = acc[r] + sh_c[CB_FCB + col];
          base[((size_t)b * da.T + t) * 512 + (col & 511)] = v;
        }
      }
    }
    // --- gate ---
    if (tid < DEC_B) {
      float s = gb;
      for (int j = 0; j < DMODEL; ++j) s += sh_h[tid * DMODEL + j] * sh_c[CB_GATE + j];
      da.gate[(size_t)t * DEC_B + tid] = 1.0f / (1.0f + __expf(-s));
    }
    __syncthreads();
  }
}

// ---------------------------------------------------------------------------
// host side
// ---------------------------------------------------------------------------

extern "C" void kernel_launch(void* const* d_in, const int* in_sizes, int n_in,
                              void* d_out, int out_size, void* d_ws, size_t ws_size,
                              hipStream_t stream) {
  (void)in_sizes; (void)n_in; (void)out_size; (void)ws_size;

  // ---- unpack inputs (setup_inputs insertion order, depth-first) ----
  int ii = 0;
  const int* src = (const int*)d_in[ii++];
  ii++; // mel_l (python int; value is the compile-time constant 400)
  const float* emb = (const float*)d_in[ii++];

  struct EncW { const float *wq,*bq,*wk,*bk,*wv,*bv,*wo,*bo,*g1,*b1,*l1w,*l1b,*l2w,*l2b,*g2,*b2; };
  EncW enc[2];
  for (int l = 0; l < 2; ++l) {
    const float** p = (const float**)&enc[l];
    for (int j = 0; j < 16; ++j) p[j] = (const float*)d_in[ii++];
  }
  struct DecW { const float *sawq,*sabq,*sawk,*sabk,*sawv,*sabv,*sawo,*sabo,
                            *cawq,*cabq,*cawk,*cabk,*cawv,*cabv,*cawo,*cabo,
                            *g1,*b1,*g2,*b2,*g3,*b3,*l1w,*l1b,*l2w,*l2b; };
  DecW dec[2];
  for (int l = 0; l < 2; ++l) {
    const float** p = (const float**)&dec[l];
    for (int j = 0; j < 26; ++j) p[j] = (const float*)d_in[ii++];
  }
  const float* fcw   = (const float*)d_in[ii++];
  const float* fcb   = (const float*)d_in[ii++];
  const float* gatew = (const float*)d_in[ii++];
  const float* gateb = (const float*)d_in[ii++];
  const float* micw  = (const float*)d_in[ii++];
  const float* micb  = (const float*)d_in[ii++];
  const float* logw  = (const float*)d_in[ii++];
  const float* logb  = (const float*)d_in[ii++];
  const float* melfb = (const float*)d_in[ii++];

  // ---- output layout: (out, gate, vel, pres, A, Bm) flattened ----
  float* out = (float*)d_out;
  const size_t N_OUT  = (size_t)DEC_B * TSTEPS * NMELC;   // 4,096,000
  const size_t N_GATE = (size_t)TSTEPS * DEC_B;           //    51,200
  const size_t NBT512 = (size_t)DEC_B * TSTEPS * 512;     // 26,214,400
  float* outMain = out;
  float* gateOut = out + N_OUT;
  float* velOut  = gateOut + N_GATE;
  float* presOut = velOut + NBT512;
  float* AOut    = presOut + NBT512;
  float* BmOut   = AOut + NBT512;

  // ---- workspace carve ----
  size_t wsOff = 0;
  auto wsAlloc = [&](size_t bytes) -> void* {
    void* p = (char*)d_ws + wsOff;
    wsOff += (bytes + 255) & ~(size_t)255;
    return p;
  };
  float*    xf    = (float*)   wsAlloc((size_t)ENC_ROWS * DMODEL * 4);
  _Float16* xh    = (_Float16*)wsAlloc((size_t)ENC_ROWS * DMODEL * 2);
  float*    tproj = (float*)   wsAlloc((size_t)ENC_ROWS * DMODEL * 4);
  _Float16* qp16  = (_Float16*)wsAlloc((size_t)ENC_ROWS * DMODEL * 2);
  _Float16* kp16  = (_Float16*)wsAlloc((size_t)ENC_ROWS * DMODEL * 2);
  _Float16* vp16  = (_Float16*)wsAlloc((size_t)ENC_ROWS * DMODEL * 2);
  _Float16* vT16  = (_Float16*)wsAlloc((size_t)ENC_B * NHEADC * HD * ENC_L * 2);
  float*    sc    = (float*)   wsAlloc((size_t)ENC_B * NHEADC * ENC_L * ENC_L * 4);
  _Float16* pr16  = (_Float16*)wsAlloc((size_t)ENC_B * NHEADC * ENC_L * ENC_L * 2);
  _Float16* ao16  = (_Float16*)wsAlloc((size_t)ENC_ROWS * DMODEL * 2);
  _Float16* tA16  = (_Float16*)wsAlloc((size_t)ENC_ROWS * DMODEL * 2);

  _Float16* encW16[2][6];
  for (int l = 0; l < 2; ++l)
    for (int j = 0; j < 6; ++j)
      encW16[l][j] = (_Float16*)wsAlloc((size_t)DMODEL * DMODEL * 2);
  _Float16* decW16[2][8];  // sa_wv, sa_wo, ca_wq, ca_wk, ca_wv, ca_wo, w1, w2
  for (int l = 0; l < 2; ++l)
    for (int j = 0; j < 8; ++j)
      decW16[l][j] = (_Float16*)wsAlloc((size_t)DMODEL * DMODEL * 2);
  _Float16* fcW16 = (_Float16*)wsAlloc((size_t)FC_OUTC * DMODEL * 2);
  _Float16* decK16[2], *decVT16[2];
  for (int l = 0; l < 2; ++l) {
    decK16[l]  = (_Float16*)wsAlloc((size_t)ENC_ROWS * DMODEL * 2);
    decVT16[l] = (_Float16*)wsAlloc((size_t)DEC_B * NHEADC * HD * MEM_L * 2);
  }
  _Float16* micW16 = (_Float16*)wsAlloc((size_t)NMELC * KPAD * 2);
  _Float16* logW16 = (_Float16*)wsAlloc((size_t)NMELC * KPAD * 2);
  _Float16* melT16 = (_Float16*)wsAlloc((size_t)NMELC * NSTFTC * 2);
  _Float16* lipsH  = (_Float16*)wsAlloc((size_t)EPI_ROWS * NSTFTC * 2);
  _Float16* melH   = (_Float16*)wsAlloc((size_t)EPI_ROWS * KPAD * 2);
  _Float16* out1H  = (_Float16*)wsAlloc((size_t)EPI_ROWS * KPAD * 2);

  auto pack = [&](const float* s, _Float16* d, int rows, int cols, int ld) {
    int tot = rows * ld;
    pack_f16_kernel<<<(tot + 255) / 256, 256, 0, stream>>>(s, d, rows, cols, ld);
  };
  auto gemm = [&](const _Float16* A, long long sA1, long long sA2, int lda,
                  const _Float16* W, long long sW1, long long sW2, int ldw,
                  const float* bias,
                  float* C, long long sC1, long long sC2, int ldc,
                  _Float16* Ch, long long sCh1, long long sCh2, int ldch,
                  int M, int N, int K, int H, int Z, float scale, int act) {
    int tiles  = (M >> 4) * (N >> 4);
    int blocks = (tiles + 7) >> 3;
    gemm_wmma<<<dim3(blocks, 1, Z), 256, 0, stream>>>(
        A, sA1, sA2, lda, W, sW1, sW2, ldw, bias,
        C, sC1, sC2, ldc, Ch, sCh1, sCh2, ldch, M, N, K, H, scale, act);
  };

  // ---- pack all GEMM weights to f16 ----
  for (int l = 0; l < 2; ++l) {
    pack(enc[l].wq,  encW16[l][0], 128, 128, 128);
    pack(enc[l].wk,  encW16[l][1], 128, 128, 128);
    pack(enc[l].wv,  encW16[l][2], 128, 128, 128);
    pack(enc[l].wo,  encW16[l][3], 128, 128, 128);
    pack(enc[l].l1w, encW16[l][4], 128, 128, 128);
    pack(enc[l].l2w, encW16[l][5], 128, 128, 128);
    pack(dec[l].sawv, decW16[l][0], 128, 128, 128);
    pack(dec[l].sawo, decW16[l][1], 128, 128, 128);
    pack(dec[l].cawq, decW16[l][2], 128, 128, 128);
    pack(dec[l].cawk, decW16[l][3], 128, 128, 128);
    pack(dec[l].cawv, decW16[l][4], 128, 128, 128);
    pack(dec[l].cawo, decW16[l][5], 128, 128, 128);
    pack(dec[l].l1w,  decW16[l][6], 128, 128, 128);
    pack(dec[l].l2w,  decW16[l][7], 128, 128, 128);
  }
  pack(fcw, fcW16, FC_OUTC, 128, 128);
  pack(micw, micW16, NMELC, NMELC, KPAD);   // zero-padded K: 80 -> 96
  pack(logw, logW16, NMELC, NMELC, KPAD);
  pack_melT_kernel<<<(NMELC * NSTFTC + 255) / 256, 256, 0, stream>>>(melfb, melT16);

  // ---- embedding + positional encoding ----
  embed_pe_kernel<<<(ENC_ROWS * DMODEL + 255) / 256, 256, 0, stream>>>(src, emb, xf, xh);

  // ---- encoder (2 layers) ----
  for (int l = 0; l < 2; ++l) {
    gemm(xh,0,0,128, encW16[l][0],0,0,128, enc[l].bq, nullptr,0,0,0, qp16,0,0,128,
         ENC_ROWS,128,128, 1,1, 1.0f,0);
    gemm(xh,0,0,128, encW16[l][1],0,0,128, enc[l].bk, nullptr,0,0,0, kp16,0,0,128,
         ENC_ROWS,128,128, 1,1, 1.0f,0);
    gemm(xh,0,0,128, encW16[l][2],0,0,128, enc[l].bv, nullptr,0,0,0, vp16,0,0,128,
         ENC_ROWS,128,128, 1,1, 1.0f,0);
    transpose_vT_kernel<<<(ENC_B*NHEADC*HD*ENC_L + 255)/256, 256, 0, stream>>>(vp16, vT16);
    // scores = (1/8) q @ k^T ; batched over (b,h)
    gemm(qp16,16384,64,128, kp16,16384,64,128, nullptr, sc,32768,16384,128,
         nullptr,0,0,0, 128,128,64, 2,ENC_B*NHEADC, 0.125f,0);
    softmax_rows128_kernel<<<(ENC_B*NHEADC*ENC_L + 7)/8, 256, 0, stream>>>(sc, pr16, ENC_B*NHEADC*ENC_L);
    // attn out = probs @ V
    gemm(pr16,32768,16384,128, vT16,16384,8192,128, nullptr, nullptr,0,0,0,
         ao16,16384,64,128, 128,64,128, 2,ENC_B*NHEADC, 1.0f,0);
    gemm(ao16,0,0,128, encW16[l][3],0,0,128, enc[l].bo, tproj,0,0,128, nullptr,0,0,0,
         ENC_ROWS,128,128, 1,1, 1.0f,0);
    resid_ln_kernel<<<ENC_ROWS/8, 256, 0, stream>>>(xf, tproj, enc[l].g1, enc[l].b1, xh, ENC_ROWS);
    gemm(xh,0,0,128, encW16[l][4],0,0,128, enc[l].l1b, nullptr,0,0,0, tA16,0,0,128,
         ENC_ROWS,128,128, 1,1, 1.0f,1);
    gemm(tA16,0,0,128, encW16[l][5],0,0,128, enc[l].l2b, tproj,0,0,128, nullptr,0,0,0,
         ENC_ROWS,128,128, 1,1, 1.0f,0);
    resid_ln_kernel<<<ENC_ROWS/8, 256, 0, stream>>>(xf, tproj, enc[l].g2, enc[l].b2, xh, ENC_ROWS);
  }

  // ---- hoisted decoder cross-attn K/V projections (loop-invariant!) ----
  for (int l = 0; l < 2; ++l) {
    gemm(xh,0,0,128, decW16[l][3],0,0,128, dec[l].cabk, nullptr,0,0,0, decK16[l],0,0,128,
         ENC_ROWS,128,128, 1,1, 1.0f,0);
    gemm(xh,0,0,128, decW16[l][4],0,0,128, dec[l].cabv, nullptr,0,0,0, vp16,0,0,128,
         ENC_ROWS,128,128, 1,1, 1.0f,0);
    transpose_caVT_kernel<<<(DEC_B*NHEADC*HD*MEM_L + 255)/256, 256, 0, stream>>>(vp16, decVT16[l]);
  }

  // ---- fused persistent decoder scan (one workgroup, 32 waves, LDS state) ----
  DecArgs da;
  for (int l = 0; l < 2; ++l) {
    da.L[l].sa_wv = decW16[l][0]; da.L[l].sa_wo = decW16[l][1];
    da.L[l].ca_wq = decW16[l][2]; da.L[l].ca_wo = decW16[l][5];
    da.L[l].w1 = decW16[l][6];    da.L[l].w2 = decW16[l][7];
    da.L[l].sa_bv = dec[l].sabv;  da.L[l].sa_bo = dec[l].sabo;
    da.L[l].ca_bq = dec[l].cabq;  da.L[l].ca_bo = dec[l].cabo;
    da.L[l].b1 = dec[l].l1b;      da.L[l].b2 = dec[l].l2b;
    da.L[l].g1 = dec[l].g1; da.L[l].be1 = dec[l].b1;
    da.L[l].g2 = dec[l].g2; da.L[l].be2 = dec[l].b2;
    da.L[l].g3 = dec[l].g3; da.L[l].be3 = dec[l].b3;
    da.L[l].caK = decK16[l]; da.L[l].caVT = decVT16[l];
  }
  da.fcW = fcW16; da.fcB = fcb; da.gateW = gatew; da.gateB = gateb;
  da.pres = presOut; da.vel = velOut; da.Am = AOut; da.Bm = BmOut;
  da.gate = gateOut; da.T = TSTEPS;

  size_t ldsBytes = (size_t)16384 * 4 * 2 + (size_t)16384 * 2 * 5 + 5376 * 4;  // ~316 KB
  (void)hipFuncSetAttribute((const void*)decoder_scan,
                            hipFuncAttributeMaxDynamicSharedMemorySize, (int)ldsBytes);
  decoder_scan<<<1, 1024, ldsBytes, stream>>>(da);

  // ---- epilogue: lips -> mel -> mic -> log ----
  lips_kernel<<<((int)NBT512 + 255) / 256, 256, 0, stream>>>(presOut, velOut, AOut, BmOut,
                                                             lipsH, (int)NBT512);
  zero_f16_kernel<<<(EPI_ROWS * KPAD + 255) / 256, 256, 0, stream>>>(melH, EPI_ROWS * KPAD);
  zero_f16_kernel<<<(EPI_ROWS * KPAD + 255) / 256, 256, 0, stream>>>(out1H, EPI_ROWS * KPAD);
  // mel = lips @ mel_fb   (W = mel_fb^T, (80,512))
  gemm(lipsH,0,0,NSTFTC, melT16,0,0,NSTFTC, nullptr, nullptr,0,0,0, melH,0,0,KPAD,
       EPI_ROWS, NMELC, NSTFTC, 1,1, 1.0f,0);
  // out1 = mel @ mic_w^T + mic_b   (K padded 80->96 with zeros)
  gemm(melH,0,0,KPAD, micW16,0,0,KPAD, micb, nullptr,0,0,0, out1H,0,0,KPAD,
       EPI_ROWS, NMELC, KPAD, 1,1, 1.0f,0);
  // out = out1 @ log_w^T + log_b  -> final output region
  gemm(out1H,0,0,KPAD, logW16,0,0,KPAD, logb, outMain,0,0,NMELC, nullptr,0,0,0,
       EPI_ROWS, NMELC, KPAD, 1,1, 1.0f,0);
}
